// build_model_89593017795335
// MI455X (gfx1250) — compile-verified, matrix operand order checked by reading, not run on hardware
//
#include <hip/hip_runtime.h>

// Problem dims (compile-time)
#define BB   64      // batch
#define TH_  256     // history length
#define TR_  256     // rollout length
#define IN_  1024    // input size
#define HH   2048    // hidden size
#define G3   6144    // 3*H
#define ROWS 16448   // (TR_+1)*BB  rows of precomputed gi (row = t*64 + b)
#define KC   1024    // k-chunk staged in LDS per step-kernel iteration
#define APAD 8       // LDS row padding (bf16 elems) to spread banks

typedef __bf16 bf16;
typedef __attribute__((ext_vector_type(16))) __bf16 v16bf;
typedef __attribute__((ext_vector_type(8)))  __bf16 v8bf;
typedef __attribute__((ext_vector_type(8)))  float  v8f;

// ---------------------------------------------------------------------------
// Fragment helpers.
// 16-bit A-matrix 16x32 layout (ISA 7.12.2): lane l holds row m = l&15,
// k-half = l>>4; the 16 bf16 values are two contiguous runs of 8:
//   k = k0 + khalf*8 .. +7   and   k = k0 + 16 + khalf*8 .. +7
// B-matrix 32x16 mirrors this on the n index, so the same global loader
// serves W rows (weights stored [n, k] row-major).
// ---------------------------------------------------------------------------
__device__ __forceinline__ v16bf combine16(v8bf a, v8bf b) {
  return __builtin_shufflevector(a, b, 0, 1, 2, 3, 4, 5, 6, 7,
                                       8, 9, 10, 11, 12, 13, 14, 15);
}

__device__ __forceinline__ v16bf load_frag(const bf16* __restrict__ row,
                                           int k0, int khalf) {
  const v8bf a = *reinterpret_cast<const v8bf*>(row + k0 + khalf * 8);
  const v8bf b = *reinterpret_cast<const v8bf*>(row + k0 + 16 + khalf * 8);
  return combine16(a, b);
}

__device__ __forceinline__ v16bf lds_frag(const bf16* sh, int base,
                                          int kk, int khalf) {
  const v8bf a = *reinterpret_cast<const v8bf*>(&sh[base + kk + khalf * 8]);
  const v8bf b = *reinterpret_cast<const v8bf*>(&sh[base + kk + 16 + khalf * 8]);
  return combine16(a, b);
}

// CDNA5 async global->LDS copy (ASYNCcnt-tracked, no VGPR data movement).
// VDST = per-lane LDS byte address (low 32 bits of the generic shared
// pointer == LDS offset per ISA 10.2 aperture mapping), VADDR = 64-bit
// global address.
__device__ __forceinline__ void async_g2l_b128(const void* gaddr,
                                               void* laddr) {
  unsigned lds_off = (unsigned)(unsigned long long)laddr;
  asm volatile("global_load_async_to_lds_b128 %0, %1, off"
               :
               : "v"(lds_off), "v"(gaddr)
               : "memory");
}

__device__ __forceinline__ void wait_async_zero() {
  asm volatile("s_wait_asynccnt 0x0" ::: "memory");
}

__device__ __forceinline__ float sigmoidf_(float x) {
  return 1.0f / (1.0f + __expf(-x));
}

#define WMMA_BF16(A, Bm, C) \
  __builtin_amdgcn_wmma_f32_16x16x32_bf16(false, (A), false, (Bm), (short)0, (C), false, false)

// ---------------------------------------------------------------------------
// Generic fp32 -> bf16 convert (used for W_ih, W_hh).
// ---------------------------------------------------------------------------
__global__ __launch_bounds__(256) void convert_bf16_kernel(
    const float* __restrict__ src, bf16* __restrict__ dst, long n) {
  long i = (long)blockIdx.x * blockDim.x + threadIdx.x;
  if (i < n) dst[i] = (bf16)src[i];
}

// ---------------------------------------------------------------------------
// Build X rows: row 0..63 = history_a_list[b, TH-1, :] (the only history step
// that matters), rows t*64+b (t>=1) = a_list[b, t-1, :].
// ---------------------------------------------------------------------------
__global__ __launch_bounds__(256) void pack_x_kernel(
    const float* __restrict__ hist_a,   // [B, TH, IN]
    const float* __restrict__ a_list,   // [B, TR, IN]
    bf16* __restrict__ xB) {            // [ROWS, IN]
  long i = (long)blockIdx.x * blockDim.x + threadIdx.x;
  if (i >= (long)ROWS * IN_) return;
  int j   = (int)(i % IN_);
  int row = (int)(i / IN_);
  int b = row & 63;
  int t = row >> 6;
  float v;
  if (t == 0) {
    v = hist_a[(long)b * TH_ * IN_ + (long)(TH_ - 1) * IN_ + j];
  } else {
    v = a_list[(long)b * TR_ * IN_ + (long)(t - 1) * IN_ + j];
  }
  xB[i] = (bf16)v;
}

// h0 = history_s_list[:, TH-1, :]  (fp32 + bf16 copies)
__global__ __launch_bounds__(256) void pack_h0_kernel(
    const float* __restrict__ hist_s,   // [B, TH, H]
    float* __restrict__ h0F, bf16* __restrict__ h0B) {
  int i = blockIdx.x * blockDim.x + threadIdx.x;
  if (i >= BB * HH) return;
  int b = i / HH, c = i % HH;
  float v = hist_s[(long)b * TH_ * HH + (long)(TH_ - 1) * HH + c];
  h0F[i] = v;
  h0B[i] = (bf16)v;
}

// ---------------------------------------------------------------------------
// GI = X @ W_ih^T : [16448 x 6144], K = 1024, bf16 WMMA.
// Block = one 16-row M-tile shared by 8 waves; A slab (16 x 1024 bf16 = 32KB)
// async-staged in LDS once.  Each wave owns a 16x32 strip (2 N-tiles) to keep
// live VGPRs well under 256 (no VGPR-MSB bank traffic).  2x-unrolled k-loop
// with SPLIT accumulators (even/odd k-halves): every accumulator has exactly
// one WMMA per iteration -> stable D==C tie; partials summed after the loop.
// Grid: 1028 mtiles x 24 wave-groups = 24672 blocks of 256 threads.
// ---------------------------------------------------------------------------
__global__ __launch_bounds__(256) void gi_gemm_kernel(
    const bf16* __restrict__ xB,    // [ROWS, IN]
    const bf16* __restrict__ wihB,  // [G3, IN]
    float* __restrict__ gi) {       // [ROWS, G3]
  __shared__ bf16 shA[16 * (IN_ + APAD)];

  const int lane  = threadIdx.x & 31;
  const int wave  = threadIdx.x >> 5;
  const int mt    = blockIdx.x % 1028;       // M-tile (shared by block)
  const int g     = blockIdx.x / 1028;       // 0..23
  const int ntg   = g * 8 + wave;            // 0..191 (pair of n-tiles)
  const int khalf = lane >> 4;
  const int nl    = lane & 15;

  // Async stage of the A slab: 16 rows x 1024 cols, 16B per lane-op.
  for (int idx = threadIdx.x; idx < 16 * (IN_ / 8); idx += 256) {
    int r  = idx >> 7;                       // IN_/8 = 128
    int c8 = idx & 127;
    async_g2l_b128(&xB[(long)(mt * 16 + r) * IN_ + c8 * 8],
                   &shA[r * (IN_ + APAD) + c8 * 8]);
  }
  wait_async_zero();
  __syncthreads();

  const bf16* b0 = wihB + (long)(ntg * 32 +  0 + nl) * IN_;
  const bf16* b1 = wihB + (long)(ntg * 32 + 16 + nl) * IN_;

  v8f e0 = {}, e1 = {};                      // even-half partials
  v8f o0 = {}, o1 = {};                      // odd-half partials
  const int abase = nl * (IN_ + APAD);

  v16bf f0 = load_frag(b0, 0, khalf);        // even-half fragments
  v16bf f1 = load_frag(b1, 0, khalf);
  v16bf g0 = load_frag(b0, 32, khalf);       // odd-half fragments
  v16bf g1 = load_frag(b1, 32, khalf);

  int kk = 0;
  for (; kk + 64 < IN_; kk += 64) {
    v16bf a = lds_frag(shA, abase, kk, khalf);
    e0 = WMMA_BF16(a, f0, e0);
    e1 = WMMA_BF16(a, f1, e1);
    f0 = load_frag(b0, kk + 64, khalf);
    f1 = load_frag(b1, kk + 64, khalf);
    v16bf a2 = lds_frag(shA, abase, kk + 32, khalf);
    o0 = WMMA_BF16(a2, g0, o0);
    o1 = WMMA_BF16(a2, g1, o1);
    g0 = load_frag(b0, kk + 96, khalf);
    g1 = load_frag(b1, kk + 96, khalf);
  }
  {  // tail: kk == IN_-64, fragments already loaded
    v16bf a = lds_frag(shA, abase, kk, khalf);
    e0 = WMMA_BF16(a, f0, e0);
    e1 = WMMA_BF16(a, f1, e1);
    v16bf a2 = lds_frag(shA, abase, kk + 32, khalf);
    o0 = WMMA_BF16(a2, g0, o0);
    o1 = WMMA_BF16(a2, g1, o1);
  }
  v8f acc0 = e0 + o0;
  v8f acc1 = e1 + o1;

  // C/D layout: lane -> n = lane&15, vgpr v -> m = v + 8*(lane>>4)
#pragma unroll
  for (int v = 0; v < 8; ++v) {
    int m   = v + 8 * khalf;
    long rw = (long)(mt * 16 + m) * G3;
    gi[rw + ntg * 32 +  0 + nl] = acc0[v];
    gi[rw + ntg * 32 + 16 + nl] = acc1[v];
  }
}

// ---------------------------------------------------------------------------
// One GRU step: gh = h_prev @ W_hh^T via WMMA, then gate math.
// Block = one batch M-tile (mt = blockIdx&3) x 8 hidden N-tiles; the block's
// 16 h_prev rows are async-staged into LDS in two 16x1024 chunks.
// Each wave accumulates r/z/n gate tiles for its (mt, nt); split even/odd
// accumulators (summed at the end) keep every WMMA chain loop-stable.
// Grid: 64 blocks x 256 threads (512 waves).
// ---------------------------------------------------------------------------
__global__ __launch_bounds__(256) void gru_step_kernel(
    int t,
    const float* __restrict__ gi,      // [ROWS, G3]
    const bf16*  __restrict__ whhB,    // [G3, H]
    const float* __restrict__ b_ih,    // [G3]
    const float* __restrict__ b_hh,    // [G3]
    const float* __restrict__ hprevF,  // [B, H]
    const bf16*  __restrict__ hprevB,  // [B, H]
    float*       __restrict__ hnewF,   // [B, H]
    bf16*        __restrict__ hnewB,   // [B, H]
    float*       __restrict__ s_out)   // s_list region of d_out: [B, TR, H]
{
  __shared__ bf16 shA[16 * (KC + APAD)];

  const int lane  = threadIdx.x & 31;
  const int wave  = threadIdx.x >> 5;
  const int mt    = blockIdx.x & 3;          // batch tile (shared by block)
  const int ntg   = blockIdx.x >> 2;         // 0..15
  const int nt    = ntg * 8 + wave;          // 0..127
  const int khalf = lane >> 4;
  const int nl    = lane & 15;
  const int mb    = mt * 16;

  const bf16* brR = whhB + (long)(0 * HH + nt * 16 + nl) * HH;
  const bf16* brZ = whhB + (long)(1 * HH + nt * 16 + nl) * HH;
  const bf16* brN = whhB + (long)(2 * HH + nt * 16 + nl) * HH;

  v8f eR = {}, eZ = {}, eN = {};             // even-half partials
  v8f oR = {}, oZ = {}, oN = {};             // odd-half partials
  const int abase = nl * (KC + APAD);

  for (int kc = 0; kc < HH; kc += KC) {
    __syncthreads();
    // Async stage of this A chunk: 16 rows x KC cols of h_prev (bf16).
    for (int idx = threadIdx.x; idx < 16 * (KC / 8); idx += 256) {
      int r  = idx >> 7;                     // KC/8 = 128
      int c8 = idx & 127;
      async_g2l_b128(&hprevB[(long)(mb + r) * HH + kc + c8 * 8],
                     &shA[r * (KC + APAD) + c8 * 8]);
    }
    // Prefetch the next chunk of this wave's B rows while we work.
    if (kc + KC < HH) {
      __builtin_prefetch(brR + kc + KC + khalf * 512, 0, 1);
      __builtin_prefetch(brZ + kc + KC + khalf * 512, 0, 1);
      __builtin_prefetch(brN + kc + KC + khalf * 512, 0, 1);
    }
    wait_async_zero();
    __syncthreads();

    v16bf fR = load_frag(brR, kc, khalf);          // even-half fragments
    v16bf fZ = load_frag(brZ, kc, khalf);
    v16bf fN = load_frag(brN, kc, khalf);
    v16bf gR = load_frag(brR, kc + 32, khalf);     // odd-half fragments
    v16bf gZ = load_frag(brZ, kc + 32, khalf);
    v16bf gN = load_frag(brN, kc + 32, khalf);

    int kk = 0;
    for (; kk + 64 < KC; kk += 64) {
      v16bf a = lds_frag(shA, abase, kk, khalf);
      eR = WMMA_BF16(a, fR, eR);
      eZ = WMMA_BF16(a, fZ, eZ);
      eN = WMMA_BF16(a, fN, eN);
      fR = load_frag(brR, kc + kk + 64, khalf);
      fZ = load_frag(brZ, kc + kk + 64, khalf);
      fN = load_frag(brN, kc + kk + 64, khalf);
      v16bf a2 = lds_frag(shA, abase, kk + 32, khalf);
      oR = WMMA_BF16(a2, gR, oR);
      oZ = WMMA_BF16(a2, gZ, oZ);
      oN = WMMA_BF16(a2, gN, oN);
      gR = load_frag(brR, kc + kk + 96, khalf);
      gZ = load_frag(brZ, kc + kk + 96, khalf);
      gN = load_frag(brN, kc + kk + 96, khalf);
    }
    {  // tail: kk == KC-64, fragments already loaded
      v16bf a = lds_frag(shA, abase, kk, khalf);
      eR = WMMA_BF16(a, fR, eR);
      eZ = WMMA_BF16(a, fZ, eZ);
      eN = WMMA_BF16(a, fN, eN);
      v16bf a2 = lds_frag(shA, abase, kk + 32, khalf);
      oR = WMMA_BF16(a2, gR, oR);
      oZ = WMMA_BF16(a2, gZ, oZ);
      oN = WMMA_BF16(a2, gN, oN);
    }
  }

  v8f accR = eR + oR;
  v8f accZ = eZ + oZ;
  v8f accN = eN + oN;

  // Gate epilogue. Column index is fixed per lane; row varies with vgpr v.
  const int cr = nt * 16 + nl;              // hidden column j in [0, 2048)
  const float bir = b_ih[cr], biz = b_ih[cr + HH], bin = b_ih[cr + 2 * HH];
  const float bhr = b_hh[cr], bhz = b_hh[cr + HH], bhn = b_hh[cr + 2 * HH];

#pragma unroll
  for (int v = 0; v < 8; ++v) {
    int m  = v + 8 * khalf;
    int bb = mb + m;                         // batch index
    const float* girow = gi + (long)(t * 64 + bb) * G3;
    float ghr = accR[v] + bhr;
    float ghz = accZ[v] + bhz;
    float ghn = accN[v] + bhn;
    float r = sigmoidf_(girow[cr]          + bir + ghr);
    float z = sigmoidf_(girow[cr + HH]     + biz + ghz);
    float n = tanhf    (girow[cr + 2 * HH] + bin + r * ghn);
    float hp = hprevF[(long)bb * HH + cr];
    float h  = (1.0f - z) * n + z * hp;
    hnewF[(long)bb * HH + cr] = h;
    hnewB[(long)bb * HH + cr] = (bf16)h;
    if (t >= 1) {
      s_out[(long)bb * (TR_ * HH) + (long)(t - 1) * HH + cr] = h;
    }
  }
}

// ---------------------------------------------------------------------------
// r_list = sigmoid(s_list @ W_r^T + b_r): one wave per (b, t) row,
// wave32 shuffle reduction. Reads h from the already-written s_list output.
// ---------------------------------------------------------------------------
__global__ __launch_bounds__(256) void reward_kernel(
    const float* __restrict__ s_out,   // [B, TR, H]
    const float* __restrict__ W_r,     // [1, H]
    const float* __restrict__ b_r,     // [1]
    float* __restrict__ r_out) {       // [B, TR]
  const int lane = threadIdx.x & 31;
  const int w    = blockIdx.x * 8 + (threadIdx.x >> 5);   // 0..16383
  const int b    = w >> 8;
  const int t    = w & 255;
  const float* h = s_out + (long)b * (TR_ * HH) + (long)t * HH;
  float s = 0.0f;
  for (int j = lane; j < HH; j += 32) s += h[j] * W_r[j];
#pragma unroll
  for (int off = 16; off > 0; off >>= 1) s += __shfl_down(s, off, 32);
  if (lane == 0) r_out[b * TR_ + t] = sigmoidf_(s + b_r[0]);
}

// ---------------------------------------------------------------------------
// Host side
// ---------------------------------------------------------------------------
extern "C" void kernel_launch(void* const* d_in, const int* in_sizes, int n_in,
                              void* d_out, int out_size, void* d_ws, size_t ws_size,
                              hipStream_t stream) {
  const float* hist_s = (const float*)d_in[0];   // [B, TH, H]
  const float* hist_a = (const float*)d_in[1];   // [B, TH, IN]
  /* d_in[2] = s, unused by the reference math */
  const float* a_list = (const float*)d_in[3];   // [B, TR, IN]
  const float* W_ih   = (const float*)d_in[4];   // [3H, IN]
  const float* W_hh   = (const float*)d_in[5];   // [3H, H]
  const float* b_ih   = (const float*)d_in[6];   // [3H]
  const float* b_hh   = (const float*)d_in[7];   // [3H]
  const float* W_r    = (const float*)d_in[8];   // [1, H]
  const float* b_r    = (const float*)d_in[9];   // [1]
  (void)in_sizes; (void)n_in; (void)out_size; (void)ws_size;

  // Workspace carve-up (256B aligned slices).
  char* ws = (char*)d_ws;
  size_t off = 0;
  auto take = [&](size_t bytes) -> char* {
    char* p = ws + off;
    off = (off + bytes + 255) & ~(size_t)255;
    return p;
  };
  bf16*  xB    = (bf16*) take((size_t)ROWS * IN_ * sizeof(bf16));
  bf16*  wihB  = (bf16*) take((size_t)G3 * IN_ * sizeof(bf16));
  bf16*  whhB  = (bf16*) take((size_t)G3 * HH * sizeof(bf16));
  float* gi    = (float*)take((size_t)ROWS * G3 * sizeof(float));
  float* h0F   = (float*)take((size_t)BB * HH * sizeof(float));
  bf16*  h0B   = (bf16*) take((size_t)BB * HH * sizeof(bf16));
  float* bufF[2];
  bf16*  bufB[2];
  bufF[0] = (float*)take((size_t)BB * HH * sizeof(float));
  bufF[1] = (float*)take((size_t)BB * HH * sizeof(float));
  bufB[0] = (bf16*) take((size_t)BB * HH * sizeof(bf16));
  bufB[1] = (bf16*) take((size_t)BB * HH * sizeof(bf16));

  float* r_out = (float*)d_out;              // [B, TR, 1]
  float* s_out = r_out + (size_t)BB * TR_;   // [B, TR, H]

  // 1) Precision conversion / packing.
  {
    long n = (long)G3 * IN_;
    convert_bf16_kernel<<<(int)((n + 255) / 256), 256, 0, stream>>>(W_ih, wihB, n);
  }
  {
    long n = (long)G3 * HH;
    convert_bf16_kernel<<<(int)((n + 255) / 256), 256, 0, stream>>>(W_hh, whhB, n);
  }
  {
    long n = (long)ROWS * IN_;
    pack_x_kernel<<<(int)((n + 255) / 256), 256, 0, stream>>>(hist_a, a_list, xB);
  }
  pack_h0_kernel<<<(BB * HH + 255) / 256, 256, 0, stream>>>(hist_s, h0F, h0B);

  // 2) Big parallel GEMM: gi for the history step + all rollout steps.
  gi_gemm_kernel<<<24672, 256, 0, stream>>>(xB, wihB, gi);

  // 3) Sequential recurrence: t=0 is the history step, t=1..TR are rollout.
  for (int t = 0; t <= TR_; ++t) {
    const float* hpF = (t == 0) ? h0F : bufF[(t - 1) & 1];
    const bf16*  hpB = (t == 0) ? h0B : bufB[(t - 1) & 1];
    gru_step_kernel<<<64, 256, 0, stream>>>(t, gi, whhB, b_ih, b_hh,
                                            hpF, hpB,
                                            bufF[t & 1], bufB[t & 1], s_out);
  }

  // 4) Reward head over all stored states.
  reward_kernel<<<2048, 256, 0, stream>>>(s_out, W_r, b_r, r_out);
}
